// Decoder_24275155157373
// MI455X (gfx1250) — matrix operand order repaired and checked by reading, not compiled, not run on hardware
//
#include <hip/hip_runtime.h>

#define B_   256
#define T_   200
#define E_   584
#define H_   8
#define HS_  73
#define FF_  2336
#define NR_  (B_*T_)        /* 51200 rows */
#define EP_  608            /* E padded to 19*32 */
#define HSP_ 96             /* HS padded to 3*32 */
#define QLD_ (H_*HSP_)      /* 768 */
#define NPE_ 640            /* 584 padded to mult of 128 (packed-weight cols) */
#define NPFF_ 2432          /* 2336 padded to mult of 128 */
#define ATT_SCALE 0.0413803f /* 584^-0.5 (reference scales by EMBED_DIM) */

typedef float  v8f   __attribute__((ext_vector_type(8)));
typedef __bf16 v16bf __attribute__((ext_vector_type(16)));

union FragU { v16bf v; uint4 q[2]; };

__device__ __forceinline__ v8f wmma_bf16(const FragU& a, const FragU& b, v8f c) {
  return __builtin_amdgcn_wmma_f32_16x16x32_bf16(false, a.v, false, b.v, (short)0, c, false, false);
}

// CDNA5 async DMA: global -> LDS, 16B per lane, tracked by ASYNCcnt.
__device__ __forceinline__ void async_b128(unsigned lds_off, unsigned long long gaddr) {
  asm volatile("global_load_async_to_lds_b128 %0, %1, off"
               :: "v"(lds_off), "v"(gaddr) : "memory");
}
__device__ __forceinline__ void wait_async0() {
  asm volatile("s_wait_asynccnt 0x0" ::: "memory");
}

// ---------------------------------------------------------------------------
// bf16 WMMA GEMM: C[M,N] = A[M,K] * B[K,N] (+bias)(+resid)(relu)
//  A row-major, lda = padded K (mult 32). M % 128 == 0.
//  B given PRE-TRANSPOSED: Bt[col][k], ldk = padded K; cols padded to 128.
//  Block 256 thr = 8 waves; block tile 128x128; wave tile 32x64 (8 wmma accs).
//  Double-buffered LDS filled by GLOBAL_LOAD_ASYNC_TO_LDS_B128 (ASYNCcnt),
//  overlapped with WMMAs on the previous buffer.
// ---------------------------------------------------------------------------
template<bool BIAS, bool RESID, bool RELU, bool OUTBF16>
__global__ __launch_bounds__(256) void gemm_wmma(
    const __bf16* __restrict__ A, const __bf16* __restrict__ Bt,
    const float* __restrict__ bias, const float* __restrict__ resid, int ldr,
    void* __restrict__ Cout, int ldc, int M, int N, int kSteps, int lda, int ldk)
{
  __shared__ __attribute__((aligned(16))) __bf16 As[2][128][40];
  __shared__ __attribute__((aligned(16))) __bf16 Bst[2][128][40];
  const int tid  = threadIdx.x;
  const int wave = tid >> 5, lane = tid & 31;
  const int cl = lane & 15, hi = lane >> 4;
  const int m0 = blockIdx.y * 128;
  const int n0 = blockIdx.x * 128;
  const int r0 = (wave & 3) * 32;       // wave's row sub-tile (2 x 16 rows)
  const int c0 = (wave >> 2) * 64;      // wave's col sub-tile (4 x 16 cols)

  // staging: 512 octs per tile, 2 per thread (oct = 8 bf16 = 16B)
  const int ro0 = tid >> 2,         co0 = (tid & 3) * 8;
  const int ro1 = (tid + 256) >> 2, co1 = ((tid + 256) & 3) * 8;

  // LDS destinations (workgroup-relative offsets) and global sources
  const unsigned bufStep = (unsigned)sizeof(As[0]);  // 10240 B per buffer
  const unsigned ldsA0 = (unsigned)(size_t)&As[0][ro0][co0];
  const unsigned ldsA1 = (unsigned)(size_t)&As[0][ro1][co1];
  const unsigned ldsB0 = (unsigned)(size_t)&Bst[0][ro0][co0];
  const unsigned ldsB1 = (unsigned)(size_t)&Bst[0][ro1][co1];
  const unsigned long long gA0 = (unsigned long long)(A  + (size_t)(m0 + ro0) * lda + co0);
  const unsigned long long gA1 = (unsigned long long)(A  + (size_t)(m0 + ro1) * lda + co1);
  const unsigned long long gB0 = (unsigned long long)(Bt + (size_t)(n0 + ro0) * ldk + co0);
  const unsigned long long gB1 = (unsigned long long)(Bt + (size_t)(n0 + ro1) * ldk + co1);

  v8f acc[2][4];
  for (int g = 0; g < 2; ++g)
    for (int j = 0; j < 4; ++j)
      for (int i = 0; i < 8; ++i) acc[g][j][i] = 0.0f;

  // ---- DMA K-slice 0 into buffer 0 ----
  async_b128(ldsA0, gA0);
  async_b128(ldsA1, gA1);
  async_b128(ldsB0, gB0);
  async_b128(ldsB1, gB1);

  for (int ks = 0; ks < kSteps; ++ks) {
    wait_async0();            // my slice-(ks) DMAs done
    __syncthreads();          // everyone's done; prior reads of other buffer done
    if (ks + 1 < kSteps) {    // kick off next slice into the other buffer
      const unsigned boff = (unsigned)((ks + 1) & 1) * bufStep;
      const unsigned long long goff = (unsigned long long)(ks + 1) * 64ull; // 32 bf16
      async_b128(ldsA0 + boff, gA0 + goff);
      async_b128(ldsA1 + boff, gA1 + goff);
      async_b128(ldsB0 + boff, gB0 + goff);
      async_b128(ldsB1 + boff, gB1 + goff);
    }
    if (ks + 2 < kSteps)      // L2 prefetch two slices ahead
      __builtin_prefetch(A + (size_t)(m0 + ro0) * lda + (ks + 2) * 32, 0, 1);

    const int pb = ks & 1;
    FragU a0, a1;
    a0.q[0] = *(const uint4*)&As[pb][r0 + cl][hi * 8];
    a0.q[1] = *(const uint4*)&As[pb][r0 + cl][16 + hi * 8];
    a1.q[0] = *(const uint4*)&As[pb][r0 + 16 + cl][hi * 8];
    a1.q[1] = *(const uint4*)&As[pb][r0 + 16 + cl][16 + hi * 8];
    #pragma unroll
    for (int j = 0; j < 4; ++j) {
      FragU b;                 // b[i] = B[K = hi*8+i (i<8), 16+hi*8+(i-8)][col]
      b.q[0] = *(const uint4*)&Bst[pb][c0 + j * 16 + cl][hi * 8];
      b.q[1] = *(const uint4*)&Bst[pb][c0 + j * 16 + cl][16 + hi * 8];
      acc[0][j] = wmma_bf16(a0, b, acc[0][j]);
      acc[1][j] = wmma_bf16(a1, b, acc[1][j]);
    }
  }

  // ---- epilogue: C frag VGPR r -> row (r + 8*hi), col = cl + 16*j ----
  #pragma unroll
  for (int g = 0; g < 2; ++g) {
    const int mrow = m0 + r0 + g * 16 + hi * 8;
    #pragma unroll
    for (int j = 0; j < 4; ++j) {
      int col = n0 + c0 + j * 16 + cl;
      if (col < N) {
        #pragma unroll
        for (int r = 0; r < 8; ++r) {
          int row = mrow + r;
          float v = acc[g][j][r];
          if (BIAS)  v += bias[col];
          if (RESID) v += resid[(size_t)row * ldr + col];
          if (RELU)  v = fmaxf(v, 0.0f);
          if (OUTBF16) ((__bf16*)Cout)[(size_t)row * ldc + col] = (__bf16)v;
          else         ((float*)Cout)[(size_t)row * ldc + col] = v;
        }
      }
    }
  }
}

// ---------------------------------------------------------------------------
// Flash-style attention. Block = 128 thr (4 waves) handles one (b,h) and 64
// query rows; loops key tiles of 32 with online softmax. Q/K/V bf16 with
// per-head stride HSP_ (pad lanes are zero). Output: concat-head bf16 [NR,EP].
// ---------------------------------------------------------------------------
template<bool CAUSAL>
__global__ __launch_bounds__(128) void attn_wmma(
    const __bf16* __restrict__ Qg, const __bf16* __restrict__ Kg,
    const __bf16* __restrict__ Vg, __bf16* __restrict__ Og)
{
  __shared__ __attribute__((aligned(16))) __bf16 Qs[64][104];
  __shared__ __attribute__((aligned(16))) __bf16 Ks[32][104];
  __shared__ __attribute__((aligned(16))) __bf16 Vst[HSP_][40];  // Vst[d][key]
  __shared__ __attribute__((aligned(16))) __bf16 Ps[4][16][40];  // per-wave P tile
  __shared__ float Ssc[4][16][32];
  __shared__ float m_s[4][16], l_s[4][16], cor_s[4][16];

  const int b = blockIdx.z, h = blockIdx.y, qt = blockIdx.x;
  const int q0 = qt * 64;
  const int tid = threadIdx.x, wave = tid >> 5, lane = tid & 31;
  const int cl = lane & 15, hi = lane >> 4;
  const size_t rowBase = (size_t)b * T_;

  if (lane < 16) { m_s[wave][lane] = -1e30f; l_s[wave][lane] = 0.0f; }

  // load Q tile 64x96 (zero pad rows >= T)
  for (int o = tid; o < 768; o += 128) {
    int r = o / 12, c8 = (o % 12) * 8;
    int t = q0 + r;
    uint4 v = make_uint4(0u, 0u, 0u, 0u);
    if (t < T_) v = *(const uint4*)(Qg + (rowBase + t) * QLD_ + h * HSP_ + c8);
    *(uint4*)&Qs[r][c8] = v;
  }

  v8f O[6];
  for (int nt = 0; nt < 6; ++nt)
    for (int i = 0; i < 8; ++i) O[nt][i] = 0.0f;

  for (int st = 0; st < 7; ++st) {
    const int s0 = st * 32;
    if (CAUSAL && s0 > q0 + 63) break;
    __syncthreads();
    // load K tile 32x96 row-major; V tile transposed into Vst[d][key]
    for (int o = tid; o < 384; o += 128) {
      int r = o / 12, c8 = (o % 12) * 8;
      int s = s0 + r;
      uint4 kv = make_uint4(0u, 0u, 0u, 0u), vv = make_uint4(0u, 0u, 0u, 0u);
      if (s < T_) {
        kv = *(const uint4*)(Kg + (rowBase + s) * QLD_ + h * HSP_ + c8);
        vv = *(const uint4*)(Vg + (rowBase + s) * QLD_ + h * HSP_ + c8);
      }
      *(uint4*)&Ks[r][c8] = kv;
      __attribute__((aligned(16))) __bf16 tmp[8];
      *(uint4*)tmp = vv;
      #pragma unroll
      for (int i = 0; i < 8; ++i) Vst[c8 + i][r] = tmp[i];
    }
    __syncthreads();

    // ---- S = Q * K^T  (wave: 16 rows x 32 keys, K-dim 96) ----
    v8f sA, sB;
    for (int i = 0; i < 8; ++i) { sA[i] = 0.0f; sB[i] = 0.0f; }
    const int rowA = wave * 16 + cl;
    #pragma unroll
    for (int kk = 0; kk < 96; kk += 32) {
      FragU a;
      a.q[0] = *(const uint4*)&Qs[rowA][kk + hi * 8];
      a.q[1] = *(const uint4*)&Qs[rowA][kk + 16 + hi * 8];
      FragU b0, b1;
      b0.q[0] = *(const uint4*)&Ks[cl][kk + hi * 8];
      b0.q[1] = *(const uint4*)&Ks[cl][kk + 16 + hi * 8];
      b1.q[0] = *(const uint4*)&Ks[cl + 16][kk + hi * 8];
      b1.q[1] = *(const uint4*)&Ks[cl + 16][kk + 16 + hi * 8];
      sA = wmma_bf16(a, b0, sA);
      sB = wmma_bf16(a, b1, sB);
    }
    #pragma unroll
    for (int r = 0; r < 8; ++r) {
      Ssc[wave][r + 8 * hi][cl]      = sA[r];
      Ssc[wave][r + 8 * hi][cl + 16] = sB[r];
    }
    __syncthreads();

    // ---- masking + online softmax (lane owns row=cl, col-half=hi) ----
    const int ro = cl;
    const int tg = q0 + wave * 16 + ro;
    float sv[16];
    float pm = -1e30f;
    #pragma unroll
    for (int c = 0; c < 16; ++c) {
      int col = hi * 16 + c, s = s0 + col;
      float v = Ssc[wave][ro][col] * ATT_SCALE;
      bool bad = (s >= T_) || (tg >= T_) || (CAUSAL && s > tg);
      v = bad ? -1e30f : v;
      sv[c] = v;
      pm = fmaxf(pm, v);
    }
    pm = fmaxf(pm, __shfl_xor(pm, 16, 32));
    float m_old = m_s[wave][ro];
    float m_new = fmaxf(m_old, pm);
    float cor = (m_old <= -5e29f) ? 0.0f : __expf(m_old - m_new);
    float psum = 0.0f;
    #pragma unroll
    for (int c = 0; c < 16; ++c) {
      float p = (sv[c] <= -5e29f || m_new <= -5e29f) ? 0.0f : __expf(sv[c] - m_new);
      Ps[wave][ro][hi * 16 + c] = (__bf16)p;
      psum += p;
    }
    psum += __shfl_xor(psum, 16, 32);
    if (hi == 0) {
      m_s[wave][ro]   = m_new;
      cor_s[wave][ro] = cor;
      l_s[wave][ro]   = l_s[wave][ro] * cor + psum;
    }
    __syncthreads();

    // ---- rescale O, then O += P * V ----
    float cr[8];
    #pragma unroll
    for (int r = 0; r < 8; ++r) cr[r] = cor_s[wave][r + 8 * hi];
    #pragma unroll
    for (int nt = 0; nt < 6; ++nt)
      #pragma unroll
      for (int r = 0; r < 8; ++r) O[nt][r] *= cr[r];

    FragU pa;
    pa.q[0] = *(const uint4*)&Ps[wave][cl][hi * 8];
    pa.q[1] = *(const uint4*)&Ps[wave][cl][16 + hi * 8];
    #pragma unroll
    for (int nt = 0; nt < 6; ++nt) {
      FragU vb;
      vb.q[0] = *(const uint4*)&Vst[nt * 16 + cl][hi * 8];
      vb.q[1] = *(const uint4*)&Vst[nt * 16 + cl][16 + hi * 8];
      O[nt] = wmma_bf16(pa, vb, O[nt]);
    }
  }
  __syncthreads();

  // ---- normalize and store (only real HS columns) ----
  float li[8];
  #pragma unroll
  for (int r = 0; r < 8; ++r) {
    float l = l_s[wave][r + 8 * hi];
    li[r] = (l > 0.0f) ? 1.0f / l : 0.0f;
  }
  #pragma unroll
  for (int nt = 0; nt < 6; ++nt) {
    int d = nt * 16 + cl;
    #pragma unroll
    for (int r = 0; r < 8; ++r) {
      int t = q0 + wave * 16 + r + 8 * hi;
      if (d < HS_ && t < T_)
        Og[(rowBase + t) * EP_ + h * HS_ + d] = (__bf16)(O[nt][r] * li[r]);
    }
  }
}

// ---------------------------------------------------------------------------
// LayerNorm: wave per row, f32 in -> bf16 out padded to EP_ (zeros in pad).
// ---------------------------------------------------------------------------
__global__ __launch_bounds__(256) void ln_rows(
    const float* __restrict__ x, const float* __restrict__ g,
    const float* __restrict__ bb, __bf16* __restrict__ out)
{
  const int wave = threadIdx.x >> 5, lane = threadIdx.x & 31;
  const int row = blockIdx.x * 8 + wave;
  const float* xr = x + (size_t)row * E_;
  float s = 0.0f, s2 = 0.0f;
  for (int e = lane; e < E_; e += 32) { float v = xr[e]; s += v; s2 += v * v; }
  #pragma unroll
  for (int m = 16; m >= 1; m >>= 1) {
    s  += __shfl_xor(s, m, 32);
    s2 += __shfl_xor(s2, m, 32);
  }
  float mean = s / (float)E_;
  float var  = s2 / (float)E_ - mean * mean;
  float rstd = rsqrtf(var + 1e-5f);
  __bf16* orow = out + (size_t)row * EP_;
  for (int e = lane; e < EP_; e += 32) {
    float v = (e < E_) ? ((xr[e] - mean) * rstd * g[e] + bb[e]) : 0.0f;
    orow[e] = (__bf16)v;
  }
}

// ---- weight repack helpers: all GEMM B-operands stored TRANSPOSED [col][k] ----
__global__ void pack_head_t(const float* __restrict__ w, __bf16* __restrict__ out) {
  int i = blockIdx.x * 256 + threadIdx.x;           // QLD_ * EP_
  if (i >= QLD_ * EP_) return;
  int c = i / EP_, k = i % EP_, h = c / HSP_, d = c % HSP_;
  out[i] = (__bf16)((k < E_ && d < HS_) ? w[((size_t)h * E_ + k) * HS_ + d] : 0.0f);
}
__global__ void pack_mat_t(const float* __restrict__ w, __bf16* __restrict__ out,
                           int K0, int N0, int Kp, int Np) {
  int i = blockIdx.x * 256 + threadIdx.x;           // Np * Kp
  if (i >= Np * Kp) return;
  int c = i / Kp, k = i % Kp;
  out[i] = (__bf16)((c < N0 && k < K0) ? w[(size_t)k * N0 + c] : 0.0f);
}
__global__ void cvt_pad(const float* __restrict__ in, __bf16* __restrict__ out) {
  size_t i = (size_t)blockIdx.x * 256 + threadIdx.x;  // NR_*EP_
  if (i >= (size_t)NR_ * EP_) return;
  size_t r = i / EP_; int e = (int)(i % EP_);
  out[i] = (__bf16)((e < E_) ? in[r * E_ + e] : 0.0f);
}
__global__ void zero_tail(__bf16* __restrict__ att) {
  int i = blockIdx.x * 256 + threadIdx.x;             // NR_*(EP_-E_)
  if (i >= NR_ * (EP_ - E_)) return;
  int r = i / (EP_ - E_), c = E_ + i % (EP_ - E_);
  att[(size_t)r * EP_ + c] = (__bf16)0.0f;
}

// ---------------------------------------------------------------------------
extern "C" void kernel_launch(void* const* d_in, const int* in_sizes, int n_in,
                              void* d_out, int out_size, void* d_ws, size_t ws_size,
                              hipStream_t stream)
{
  (void)in_sizes; (void)n_in; (void)out_size; (void)ws_size;
  const float* idx    = (const float*)d_in[0];
  const float* memory = (const float*)d_in[1];
  const float* sa_q  = (const float*)d_in[3];
  const float* sa_k  = (const float*)d_in[4];
  const float* sa_v  = (const float*)d_in[5];
  const float* sa_pw = (const float*)d_in[6];
  const float* sa_pb = (const float*)d_in[7];
  const float* ca_q  = (const float*)d_in[8];
  const float* ca_k  = (const float*)d_in[9];
  const float* ca_v  = (const float*)d_in[10];
  const float* ca_pw = (const float*)d_in[11];
  const float* ca_pb = (const float*)d_in[12];
  const float* ff_w1 = (const float*)d_in[13];
  const float* ff_b1 = (const float*)d_in[14];
  const float* ff_w2 = (const float*)d_in[15];
  const float* ff_b2 = (const float*)d_in[16];
  const float* ln1w = (const float*)d_in[17];
  const float* ln1b = (const float*)d_in[18];
  const float* ln2w = (const float*)d_in[19];
  const float* ln2b = (const float*)d_in[20];
  const float* ln3w = (const float*)d_in[21];
  const float* ln3b = (const float*)d_in[22];
  float* out = (float*)d_out;

  char* ws = (char*)d_ws;
  size_t off = 0;
  auto give = [&](size_t bytes) -> void* {
    void* p = ws + off;
    off = (off + bytes + 255) & ~(size_t)255;
    return p;
  };
  __bf16* wqs = (__bf16*)give((size_t)QLD_ * EP_ * 2);
  __bf16* wks = (__bf16*)give((size_t)QLD_ * EP_ * 2);
  __bf16* wvs = (__bf16*)give((size_t)QLD_ * EP_ * 2);
  __bf16* wqc = (__bf16*)give((size_t)QLD_ * EP_ * 2);
  __bf16* wkc = (__bf16*)give((size_t)QLD_ * EP_ * 2);
  __bf16* wvc = (__bf16*)give((size_t)QLD_ * EP_ * 2);
  __bf16* wps = (__bf16*)give((size_t)NPE_ * EP_ * 2);
  __bf16* wpc = (__bf16*)give((size_t)NPE_ * EP_ * 2);
  __bf16* w1p = (__bf16*)give((size_t)NPFF_ * EP_ * 2);
  __bf16* w2p = (__bf16*)give((size_t)NPE_ * FF_ * 2);
  __bf16* hln = (__bf16*)give((size_t)NR_ * EP_ * 2);
  __bf16* Qb  = (__bf16*)give((size_t)NR_ * QLD_ * 2);
  __bf16* Kb  = (__bf16*)give((size_t)NR_ * QLD_ * 2);
  __bf16* Vb  = (__bf16*)give((size_t)NR_ * QLD_ * 2);
  __bf16* att = (__bf16*)give((size_t)NR_ * EP_ * 2);
  __bf16* memb= (__bf16*)give((size_t)NR_ * EP_ * 2);
  float*  xa  = (float*)give((size_t)NR_ * E_ * 4);
  float*  xb  = (float*)give((size_t)NR_ * E_ * 4);
  __bf16* ff1 = (__bf16*)give((size_t)NR_ * FF_ * 2);

  const int KS_E  = EP_ / 32;   // 19
  const int KS_FF = FF_ / 32;   // 73
  dim3 gQKV(QLD_ / 128, NR_ / 128);       // (6, 400)
  dim3 gE(NPE_ / 128, NR_ / 128);         // (5, 400)
  dim3 gFF(NPFF_ / 128, NR_ / 128);       // (19, 400)
  dim3 gAtt(4, H_, B_);

  // ---- weight repacking (bf16, transposed [col][k], zero padded) ----
  pack_head_t<<<(QLD_*EP_ + 255)/256, 256, 0, stream>>>(sa_q, wqs);
  pack_head_t<<<(QLD_*EP_ + 255)/256, 256, 0, stream>>>(sa_k, wks);
  pack_head_t<<<(QLD_*EP_ + 255)/256, 256, 0, stream>>>(sa_v, wvs);
  pack_head_t<<<(QLD_*EP_ + 255)/256, 256, 0, stream>>>(ca_q, wqc);
  pack_head_t<<<(QLD_*EP_ + 255)/256, 256, 0, stream>>>(ca_k, wkc);
  pack_head_t<<<(QLD_*EP_ + 255)/256, 256, 0, stream>>>(ca_v, wvc);
  pack_mat_t<<<(NPE_*EP_ + 255)/256, 256, 0, stream>>>(sa_pw, wps, E_, E_, EP_, NPE_);
  pack_mat_t<<<(NPE_*EP_ + 255)/256, 256, 0, stream>>>(ca_pw, wpc, E_, E_, EP_, NPE_);
  pack_mat_t<<<(NPFF_*EP_ + 255)/256, 256, 0, stream>>>(ff_w1, w1p, E_, FF_, EP_, NPFF_);
  pack_mat_t<<<(NPE_*FF_ + 255)/256, 256, 0, stream>>>(ff_w2, w2p, FF_, E_, FF_, NPE_);
  cvt_pad<<<(unsigned)(((size_t)NR_*EP_ + 255)/256), 256, 0, stream>>>(memory, memb);
  zero_tail<<<(NR_*(EP_-E_) + 255)/256, 256, 0, stream>>>(att);

  // ==== masked self-attention + residual ====
  ln_rows<<<NR_/8, 256, 0, stream>>>(idx, ln1w, ln1b, hln);
  gemm_wmma<false,false,false,true><<<gQKV, 256, 0, stream>>>(hln, wqs, nullptr, nullptr, 0, Qb, QLD_, NR_, QLD_, KS_E, EP_, EP_);
  gemm_wmma<false,false,false,true><<<gQKV, 256, 0, stream>>>(hln, wks, nullptr, nullptr, 0, Kb, QLD_, NR_, QLD_, KS_E, EP_, EP_);
  gemm_wmma<false,false,false,true><<<gQKV, 256, 0, stream>>>(hln, wvs, nullptr, nullptr, 0, Vb, QLD_, NR_, QLD_, KS_E, EP_, EP_);
  attn_wmma<true><<<gAtt, 128, 0, stream>>>(Qb, Kb, Vb, att);
  gemm_wmma<true,true,false,false><<<gE, 256, 0, stream>>>(att, wps, sa_pb, idx, E_, xa, E_, NR_, E_, KS_E, EP_, EP_);

  // ==== cross-attention (q from x, k from memory, v from x) + residual ====
  ln_rows<<<NR_/8, 256, 0, stream>>>(xa, ln2w, ln2b, hln);
  gemm_wmma<false,false,false,true><<<gQKV, 256, 0, stream>>>(hln,  wqc, nullptr, nullptr, 0, Qb, QLD_, NR_, QLD_, KS_E, EP_, EP_);
  gemm_wmma<false,false,false,true><<<gQKV, 256, 0, stream>>>(memb, wkc, nullptr, nullptr, 0, Kb, QLD_, NR_, QLD_, KS_E, EP_, EP_);
  gemm_wmma<false,false,false,true><<<gQKV, 256, 0, stream>>>(hln,  wvc, nullptr, nullptr, 0, Vb, QLD_, NR_, QLD_, KS_E, EP_, EP_);
  attn_wmma<false><<<gAtt, 128, 0, stream>>>(Qb, Kb, Vb, att);
  gemm_wmma<true,true,false,false><<<gE, 256, 0, stream>>>(att, wpc, ca_pb, xa, E_, xb, E_, NR_, E_, KS_E, EP_, EP_);

  // ==== feed-forward + residual ====
  ln_rows<<<NR_/8, 256, 0, stream>>>(xb, ln3w, ln3b, hln);
  gemm_wmma<true,false,true,true><<<gFF, 256, 0, stream>>>(hln, w1p, ff_b1, nullptr, 0, ff1, FF_, NR_, FF_, KS_E, EP_, EP_);
  gemm_wmma<true,true,false,false><<<gE, 256, 0, stream>>>(ff1, w2p, ff_b2, xb, E_, out, E_, NR_, E_, KS_FF, FF_, FF_);
}
